// VAE_48833778155749
// MI455X (gfx1250) — matrix operand order repaired and checked by reading, not compiled, not run on hardware
//
#include <hip/hip_runtime.h>

// ---------------------------------------------------------------------------
// CDNA5 (gfx1250) VAE forward: bf16 WMMA LSTM steps + gathered one-hot inputs.
// ---------------------------------------------------------------------------

typedef __attribute__((ext_vector_type(16))) __bf16 v16bf;
typedef __attribute__((ext_vector_type(8)))  float  v8f;

__device__ inline float sigmoidf_(float x) { return 1.f / (1.f + __expf(-x)); }

// A-fragment (16x32 bf16, M x K) from LDS tile laid out [row][32k].
// lanes 0-15: row=lane, K {0..7,16..23}; lanes 16-31: row=lane-16, K {8..15,24..31}
__device__ inline v16bf ldfragA(const __bf16* lA, int rowbase, int lane) {
  union { v16bf v; uint4 q[2]; } u;
  int half = lane >> 4;
  const __bf16* p = lA + (rowbase + (lane & 15)) * 32 + half * 8;
  u.q[0] = *(const uint4*)p;
  u.q[1] = *(const uint4*)(p + 16);
  return u.v;
}
// B-fragment (32x16 bf16, K x N). LDS tile stores weight rows n x 32 k.
// lanes 0-15: n=lane, K {0..7,16..23}; lanes 16-31: n=lane-16, K {8..15,24..31}
__device__ inline v16bf ldfragB(const __bf16* lB, int nbase, int lane) {
  union { v16bf v; uint4 q[2]; } u;
  int half = lane >> 4;
  const __bf16* p = lB + (nbase + (lane & 15)) * 32 + half * 8;
  u.q[0] = *(const uint4*)p;
  u.q[1] = *(const uint4*)(p + 16);
  return u.v;
}
__device__ inline v8f wmma_bf(v16bf a, v16bf b, v8f c) {
  return __builtin_amdgcn_wmma_f32_16x16x32_bf16(false, a, false, b, (short)0, c,
                                                 false, false);
}

// ---------------------------------------------------------------------------
// Fused LSTM step: g = h@Whh_r^T (+ gathered x-contrib + bias), pointwise update.
// Grid (8 B-tiles, 8 CF-tiles), 128 threads = 4 waves; wave w: rows 16w..16w+15,
// all 256 reordered cols (4 gates x 64 units).  Weights gate-reordered:
// n' = 256*cb + g*64 + j  <->  orig row g*512 + cb*64 + j.
// mode 0: encoder  (ia=acts[B][64], ib=ress, fv=ts)
// mode 1: act LSTM (pre + col(516+ia[b]))
// mode 2: res LSTM (pre + col(516+ia[b]) + col(580+ib[b]))
// mode 3: ts  LSTM (pre + col(516+ia[b]) + fv[b]*col(580))
// ---------------------------------------------------------------------------
__global__ __launch_bounds__(128) void k_lstm_step(
    const __bf16* __restrict__ hin, const __bf16* __restrict__ whh,
    const float* __restrict__ bias, const float* __restrict__ wt,
    const float* __restrict__ pre, float* __restrict__ cst,
    __bf16* __restrict__ hout, const int* __restrict__ ia,
    const int* __restrict__ ib, const float* __restrict__ fv,
    int t, int mode, const int* __restrict__ lens, __bf16* __restrict__ snap)
{
  __shared__ __align__(16) __bf16 lA[64 * 32];
  __shared__ __align__(16) __bf16 lB[256 * 32];
  const int tid = threadIdx.x;
  const int lane = tid & 31;
  const int w = tid >> 5;
  const int bb = blockIdx.x, cb = blockIdx.y;

  v8f acc[16];
  const v8f vz = {0.f,0.f,0.f,0.f,0.f,0.f,0.f,0.f};
#pragma unroll
  for (int i = 0; i < 16; ++i) acc[i] = vz;

  const int rowg = 64 * bb;
  for (int kb = 0; kb < 512; kb += 32) {
#pragma unroll
    for (int j = 0; j < 2; ++j) {          // A tile: 64x32 bf16 = 256 uint4
      int i = tid + j * 128;
      int ar = i >> 2, ac8 = (i & 3) * 8;
      *(uint4*)&lA[ar * 32 + ac8] =
          *(const uint4*)&hin[(rowg + ar) * 512 + kb + ac8];
    }
#pragma unroll
    for (int j = 0; j < 8; ++j) {          // B tile: 256x32 bf16 = 1024 uint4
      int i = tid + j * 128;
      int br = i >> 2, bc8 = (i & 3) * 8;
      *(uint4*)&lB[br * 32 + bc8] =
          *(const uint4*)&whh[(256 * cb + br) * 512 + kb + bc8];
    }
    __syncthreads();
    v16bf af = ldfragA(lA, 16 * w, lane);
#pragma unroll
    for (int ctn = 0; ctn < 16; ++ctn) {
      v16bf bfr = ldfragB(lB, 16 * ctn, lane);
      acc[ctn] = wmma_bf(af, bfr, acc[ctn]);
    }
    __syncthreads();
  }

  // ---- epilogue: lane-local i/f/g/o -> c,h update ----
  const int col = lane & 15;
  const int halfm = (lane >> 4) * 8;
#pragma unroll
  for (int ct = 0; ct < 4; ++ct) {
    int ncol = 16 * ct + col;
    int n0 = 256 * cb + ncol;        // gate i col (f:+64, g:+128, o:+192)
    int u  = 64 * cb + ncol;         // CF unit
    float bi = bias[n0], bff = bias[n0 + 64];
    float bg = bias[n0 + 128], bo = bias[n0 + 192];
#pragma unroll
    for (int r = 0; r < 8; ++r) {
      int row = 64 * bb + 16 * w + halfm + r;
      float xi, xf, xg, xo;
      if (mode == 0) {
        int ac = ia[row * 64 + t], rc = ib[row * 64 + t];
        float tv = fv[row * 64 + t];
        const float* wa = wt + ac * 2048;
        const float* wtv = wt + 65 * 2048;
        const float* wr = wt + (66 + rc) * 2048;
        xi = wa[n0]       + tv * wtv[n0]       + wr[n0];
        xf = wa[n0 + 64]  + tv * wtv[n0 + 64]  + wr[n0 + 64];
        xg = wa[n0 + 128] + tv * wtv[n0 + 128] + wr[n0 + 128];
        xo = wa[n0 + 192] + tv * wtv[n0 + 192] + wr[n0 + 192];
      } else {
        const float* pr = pre + row * 2048;
        const float* wa = wt + (516 + ia[row]) * 2048;
        xi = pr[n0]       + wa[n0];
        xf = pr[n0 + 64]  + wa[n0 + 64];
        xg = pr[n0 + 128] + wa[n0 + 128];
        xo = pr[n0 + 192] + wa[n0 + 192];
        if (mode == 2) {
          const float* wr = wt + (580 + ib[row]) * 2048;
          xi += wr[n0]; xf += wr[n0 + 64]; xg += wr[n0 + 128]; xo += wr[n0 + 192];
        } else if (mode == 3) {
          float tv = fv[row];
          const float* wtv = wt + 580 * 2048;
          xi += tv * wtv[n0]; xf += tv * wtv[n0 + 64];
          xg += tv * wtv[n0 + 128]; xo += tv * wtv[n0 + 192];
        }
      }
      float iv = acc[ct][r]      + bi  + xi;
      float fvv = acc[ct + 4][r] + bff + xf;
      float gv = acc[ct + 8][r]  + bg  + xg;
      float ov = acc[ct + 12][r] + bo  + xo;
      int ci = row * 512 + u;
      float cold = cst[ci];
      float cnew = sigmoidf_(fvv) * cold + sigmoidf_(iv) * tanhf(gv);
      float hv = sigmoidf_(ov) * tanhf(cnew);
      cst[ci] = cnew;
      hout[row * 512 + u] = (__bf16)hv;
      if (snap) {
        int st = (lens[row] + 63) & 63;
        if (t == st) snap[row * 544 + 4 + u] = (__bf16)hv;
      }
    }
  }
}

// ---------------------------------------------------------------------------
// Generic bf16 WMMA GEMM: C[M][*] = A[M][KP] x B[NP][KP]^T (+bias, act).
// Grid (M/64, NP/64), 128 threads = 4 waves (16 rows x 64 cols each).
// act: 0=none, 1=relu.  C (f32) and Cb (bf16) optional.
// ---------------------------------------------------------------------------
__global__ __launch_bounds__(128) void k_gemm(
    const __bf16* __restrict__ A, int lda, const __bf16* __restrict__ Bm,
    float* __restrict__ C, int ldc, __bf16* __restrict__ Cb, int ldcb,
    const float* __restrict__ bias, int nvalid, int KP, int act)
{
  __shared__ __align__(16) __bf16 lA[64 * 32];
  __shared__ __align__(16) __bf16 lB[64 * 32];
  const int tid = threadIdx.x;
  const int lane = tid & 31;
  const int w = tid >> 5;
  const int mb = blockIdx.x, nb = blockIdx.y;

  v8f acc[4];
  const v8f vz = {0.f,0.f,0.f,0.f,0.f,0.f,0.f,0.f};
#pragma unroll
  for (int i = 0; i < 4; ++i) acc[i] = vz;

  for (int kb = 0; kb < KP; kb += 32) {
#pragma unroll
    for (int j = 0; j < 2; ++j) {
      int i = tid + j * 128;
      int r = i >> 2, c8 = (i & 3) * 8;
      *(uint4*)&lA[r * 32 + c8] = *(const uint4*)&A[(64 * mb + r) * lda + kb + c8];
      *(uint4*)&lB[r * 32 + c8] = *(const uint4*)&Bm[(64 * nb + r) * KP + kb + c8];
    }
    __syncthreads();
    v16bf af = ldfragA(lA, 16 * w, lane);
#pragma unroll
    for (int ct = 0; ct < 4; ++ct) {
      v16bf bfr = ldfragB(lB, 16 * ct, lane);
      acc[ct] = wmma_bf(af, bfr, acc[ct]);
    }
    __syncthreads();
  }

  const int colb = lane & 15;
  const int halfm = (lane >> 4) * 8;
#pragma unroll
  for (int ct = 0; ct < 4; ++ct) {
    int col = 64 * nb + 16 * ct + colb;
#pragma unroll
    for (int r = 0; r < 8; ++r) {
      int row = 64 * mb + 16 * w + halfm + r;
      if (col < nvalid) {
        float v = acc[ct][r];
        if (bias) v += bias[col];
        if (act == 1) v = v > 0.f ? v : 0.f;
        if (C)  C[row * ldc + col] = v;
        if (Cb) Cb[row * ldcb + col] = (__bf16)v;
      }
    }
  }
}

// ---------------------------------------------------------------------------
// One-time weight prep
// ---------------------------------------------------------------------------
// Gate-reorder Whh->bf16, bias, and transpose Wih to [cols][2048] (reordered n').
__global__ void k_reorder_lstm(const float* __restrict__ whh,
                               const float* __restrict__ bias,
                               const float* __restrict__ wih, int ncols,
                               __bf16* __restrict__ whh_r, float* __restrict__ bias_r,
                               float* __restrict__ wt) {
  int n = blockIdx.x * blockDim.x + threadIdx.x;
  if (n >= 2048) return;
  int cb = n >> 8, g = (n >> 6) & 3, j = n & 63;
  int orig = g * 512 + cb * 64 + j;
  for (int k = 0; k < 512; ++k) whh_r[n * 512 + k] = (__bf16)whh[orig * 512 + k];
  bias_r[n] = bias[orig];
  for (int c = 0; c < ncols; ++c) wt[c * 2048 + n] = wih[orig * ncols + c];
}

// Build padded bf16 B matrix (weight rows), optional gate-reorder, zero pad.
__global__ void k_make_b(const float* __restrict__ src, int srcN, int srcK,
                         int k0, int klen, __bf16* __restrict__ dst,
                         int NP, int KP, int reorder) {
  int idx = blockIdx.x * blockDim.x + threadIdx.x;
  if (idx >= NP * KP) return;
  int n = idx / KP, k = idx - n * KP;
  float v = 0.f;
  if (n < srcN && k < klen) {
    int row = n;
    if (reorder) { int cb = n >> 8, g = (n >> 6) & 3, j = n & 63; row = g * 512 + cb * 64 + j; }
    v = src[row * srcK + k0 + k];
  }
  dst[idx] = (__bf16)v;
}

__global__ void k_pad_bias(const float* __restrict__ src, int n,
                           float* __restrict__ dst, int NP) {
  int i = blockIdx.x * blockDim.x + threadIdx.x;
  if (i < NP) dst[i] = (i < n) ? src[i] : 0.f;
}

__global__ void k_zero(float* p, int n) {
  int i = blockIdx.x * blockDim.x + threadIdx.x;
  int st = gridDim.x * blockDim.x;
  for (; i < n; i += st) p[i] = 0.f;
}

// lens, e_attrs, initial decoder state
__global__ void k_init_misc(const int* __restrict__ attr0, const int* __restrict__ acts,
                            const float* __restrict__ a2aW, const float* __restrict__ a2ab,
                            int* __restrict__ lens, int* __restrict__ actidx,
                            int* __restrict__ residx, float* __restrict__ tsprev,
                            __bf16* __restrict__ tbuf) {
  int b = blockIdx.x * blockDim.x + threadIdx.x;
  if (b >= 512) return;
  int best = -1, bi = 0;
  for (int tt = 0; tt < 64; ++tt) {
    int v = acts[b * 64 + tt];
    if (v > best) { best = v; bi = tt; }
  }
  lens[b] = bi;
  actidx[b] = 63; residx[b] = 63; tsprev[b] = 0.f;
  int a0 = attr0[b];
  for (int i2 = 0; i2 < 4; ++i2) {
    float v = a2aW[i2 * 8 + a0] + a2ab[i2];
    tbuf[b * 544 + i2] = (__bf16)(v > 0.f ? v : 0.f);
  }
}

__global__ void k_z(const float* __restrict__ mean, const float* __restrict__ logvar,
                    const float* __restrict__ eps, __bf16* __restrict__ z) {
  int i = blockIdx.x * blockDim.x + threadIdx.x;
  if (i >= 512 * 128) return;
  z[i] = (__bf16)(mean[i] + eps[i] * __expf(0.5f * logvar[i]));
}

__global__ void k_argmax64(const float* __restrict__ base, int ldrow,
                           int* __restrict__ out) {
  __shared__ float sv[64];
  __shared__ int si[64];
  int b = blockIdx.x, tid = threadIdx.x;
  sv[tid] = base[b * ldrow + tid];
  si[tid] = tid;
  __syncthreads();
  for (int s = 32; s > 0; s >>= 1) {
    if (tid < s) {
      if (sv[tid + s] > sv[tid] ||
          (sv[tid + s] == sv[tid] && si[tid + s] < si[tid])) {
        sv[tid] = sv[tid + s]; si[tid] = si[tid + s];
      }
    }
    __syncthreads();
  }
  if (tid == 0) out[b] = si[0];
}

__global__ void k_softmax64(float* __restrict__ p) {
  __shared__ float s[64];
  int r = blockIdx.x, tid = threadIdx.x;
  float v = p[(size_t)r * 64 + tid];
  s[tid] = v;
  __syncthreads();
  for (int st = 32; st > 0; st >>= 1) {
    if (tid < st) s[tid] = fmaxf(s[tid], s[tid + st]);
    __syncthreads();
  }
  float m = s[0];
  __syncthreads();
  float e = __expf(v - m);
  s[tid] = e;
  __syncthreads();
  for (int st = 32; st > 0; st >>= 1) {
    if (tid < st) s[tid] += s[tid + st];
    __syncthreads();
  }
  p[(size_t)r * 64 + tid] = e / s[0];
}

__global__ void k_softmax8(const float* __restrict__ in, int ldin,
                           float* __restrict__ out) {
  int b = blockIdx.x;
  if (threadIdx.x != 0) return;
  float m = -1e30f;
  for (int i = 0; i < 8; ++i) m = fmaxf(m, in[b * ldin + i]);
  float s = 0.f, e[8];
  for (int i = 0; i < 8; ++i) { e[i] = __expf(in[b * ldin + i] - m); s += e[i]; }
  for (int i = 0; i < 8; ++i) out[b * 8 + i] = e[i] / s;
}

__global__ void k_ts_head2(const float* __restrict__ tsh, const float* __restrict__ w2,
                           const float* __restrict__ b2, float* __restrict__ dts,
                           float* __restrict__ tsprev, int t) {
  int b = blockIdx.x, lane = threadIdx.x;  // 32 threads
  float s = 0.f;
  for (int j = lane; j < 256; j += 32) s += tsh[b * 256 + j] * w2[j];
  for (int off = 16; off > 0; off >>= 1) s += __shfl_xor(s, off);
  if (lane == 0) {
    float v = s + b2[0];
    dts[b * 64 + t] = v;
    tsprev[b] = v;
  }
}

// ---------------------------------------------------------------------------
extern "C" void kernel_launch(void* const* d_in, const int* in_sizes, int n_in,
                              void* d_out, int out_size, void* d_ws, size_t ws_size,
                              hipStream_t stream) {
  (void)in_sizes; (void)n_in; (void)out_size; (void)ws_size;
  const int*   attr0 = (const int*)d_in[0];
  const int*   acts  = (const int*)d_in[1];
  const float* tsin  = (const float*)d_in[2];
  const int*   ress  = (const int*)d_in[3];
  const float* eps   = (const float*)d_in[4];
  const float* a2aW  = (const float*)d_in[5];
  const float* a2ab  = (const float*)d_in[6];
  const float* enc_Wih = (const float*)d_in[7];
  const float* enc_Whh = (const float*)d_in[8];
  const float* enc_b   = (const float*)d_in[9];
  const float* t2mean_W = (const float*)d_in[10];
  const float* t2mean_b = (const float*)d_in[11];
  const float* t2var_W  = (const float*)d_in[12];
  const float* t2var_b  = (const float*)d_in[13];
  const float* z2t_W    = (const float*)d_in[14];
  const float* z2t_b    = (const float*)d_in[15];
  const float* attr_W1  = (const float*)d_in[16];
  const float* attr_b1  = (const float*)d_in[17];
  const float* attr_W2  = (const float*)d_in[18];
  const float* attr_b2  = (const float*)d_in[19];
  const float* act_Wih  = (const float*)d_in[20];
  const float* act_Whh  = (const float*)d_in[21];
  const float* act_b    = (const float*)d_in[22];
  const float* res_Wih  = (const float*)d_in[23];
  const float* res_Whh  = (const float*)d_in[24];
  const float* res_b    = (const float*)d_in[25];
  const float* ts_Wih   = (const float*)d_in[26];
  const float* ts_Whh   = (const float*)d_in[27];
  const float* ts_b     = (const float*)d_in[28];
  const float* e2act_W  = (const float*)d_in[29];
  const float* e2act_b  = (const float*)d_in[30];
  const float* e2res_W  = (const float*)d_in[31];
  const float* e2res_b  = (const float*)d_in[32];
  const float* e2ts_W1  = (const float*)d_in[33];
  const float* e2ts_b1  = (const float*)d_in[34];
  const float* e2ts_W2  = (const float*)d_in[35];
  const float* e2ts_b2  = (const float*)d_in[36];

  float* fout = (float*)d_out;
  const size_t ATTR_OFF = 0, ACTS_OFF = 4096, TS_OFF = 2101248,
               RESS_OFF = 2134016, MEAN_OFF = 4231168, LOGVAR_OFF = 4296704;

  // ---- workspace bump allocator ----
  char* base = (char*)d_ws;
  size_t off = 0;
  auto alloc = [&](size_t bytes) -> void* {
    void* p = base + off;
    off = (off + bytes + 255) & ~(size_t)255;
    return p;
  };
  __bf16* WHH_ENC = (__bf16*)alloc(2048 * 512 * 2);
  __bf16* WHH_ACT = (__bf16*)alloc(2048 * 512 * 2);
  __bf16* WHH_RES = (__bf16*)alloc(2048 * 512 * 2);
  __bf16* WHH_TS  = (__bf16*)alloc(2048 * 512 * 2);
  float* BIAS_ENC = (float*)alloc(2048 * 4);
  float* BIAS_ACT = (float*)alloc(2048 * 4);
  float* BIAS_RES = (float*)alloc(2048 * 4);
  float* BIAS_TS  = (float*)alloc(2048 * 4);
  float* WT_ENC = (float*)alloc((size_t)131 * 2048 * 4);
  float* WT_ACT = (float*)alloc((size_t)580 * 2048 * 4);
  float* WT_RES = (float*)alloc((size_t)644 * 2048 * 4);
  float* WT_TS  = (float*)alloc((size_t)581 * 2048 * 4);
  __bf16* B_ACTPRE = (__bf16*)alloc((size_t)2048 * 544 * 2);
  __bf16* B_RESPRE = (__bf16*)alloc((size_t)2048 * 544 * 2);
  __bf16* B_TSPRE  = (__bf16*)alloc((size_t)2048 * 544 * 2);
  __bf16* B_T2MEAN = (__bf16*)alloc(128 * 544 * 2);
  __bf16* B_T2VAR  = (__bf16*)alloc(128 * 544 * 2);
  __bf16* B_Z2T    = (__bf16*)alloc(576 * 128 * 2);
  __bf16* B_ATTR1  = (__bf16*)alloc(320 * 544 * 2);
  __bf16* B_ATTR2  = (__bf16*)alloc(64 * 288 * 2);
  __bf16* B_E2ACT  = (__bf16*)alloc(64 * 512 * 2);
  __bf16* B_E2RES  = (__bf16*)alloc(64 * 512 * 2);
  __bf16* B_E2TS1  = (__bf16*)alloc(256 * 512 * 2);
  float* PB_T2MEAN = (float*)alloc(128 * 4);
  float* PB_T2VAR  = (float*)alloc(128 * 4);
  float* PB_Z2T    = (float*)alloc(576 * 4);
  float* PB_ATTR1  = (float*)alloc(320 * 4);
  float* PB_ATTR2  = (float*)alloc(64 * 4);
  float* PB_E2ACT  = (float*)alloc(64 * 4);
  float* PB_E2RES  = (float*)alloc(64 * 4);
  float* PB_E2TS1  = (float*)alloc(256 * 4);
  __bf16* HBUF = (__bf16*)alloc((size_t)8 * 512 * 512 * 2);  // 4 LSTMs x 2 buffers
  __bf16* H_ENC0 = HBUF + 0 * 512 * 512;
  __bf16* H_ENC1 = HBUF + 1 * 512 * 512;
  __bf16* H_ACT0 = HBUF + 2 * 512 * 512;
  __bf16* H_ACT1 = HBUF + 3 * 512 * 512;
  __bf16* H_RES0 = HBUF + 4 * 512 * 512;
  __bf16* H_RES1 = HBUF + 5 * 512 * 512;
  __bf16* H_TS0  = HBUF + 6 * 512 * 512;
  __bf16* H_TS1  = HBUF + 7 * 512 * 512;
  float* CBUF = (float*)alloc((size_t)4 * 512 * 512 * 4);
  float* C_ENC = CBUF + 0 * 512 * 512;
  float* C_ACT = CBUF + 1 * 512 * 512;
  float* C_RES = CBUF + 2 * 512 * 512;
  float* C_TS  = CBUF + 3 * 512 * 512;
  int* LENS   = (int*)alloc(512 * 4);
  int* ACTIDX = (int*)alloc(512 * 4);
  int* RESIDX = (int*)alloc(512 * 4);
  float* TSPREV = (float*)alloc(512 * 4);
  __bf16* T_BF16    = (__bf16*)alloc((size_t)512 * 544 * 2);
  __bf16* Z_BF16    = (__bf16*)alloc(512 * 128 * 2);
  __bf16* TREC_BF16 = (__bf16*)alloc((size_t)512 * 544 * 2);
  __bf16* A1_BF16   = (__bf16*)alloc((size_t)512 * 288 * 2);
  float* ATTR_TMP = (float*)alloc(512 * 64 * 4);
  float* PRE_ACT = (float*)alloc((size_t)512 * 2048 * 4);
  float* PRE_RES = (float*)alloc((size_t)512 * 2048 * 4);
  float* PRE_TS  = (float*)alloc((size_t)512 * 2048 * 4);
  float* TSH = (float*)alloc(512 * 256 * 4);

  // ---- one-time weight prep ----
  k_reorder_lstm<<<8, 256, 0, stream>>>(enc_Whh, enc_b, enc_Wih, 131, WHH_ENC, BIAS_ENC, WT_ENC);
  k_reorder_lstm<<<8, 256, 0, stream>>>(act_Whh, act_b, act_Wih, 580, WHH_ACT, BIAS_ACT, WT_ACT);
  k_reorder_lstm<<<8, 256, 0, stream>>>(res_Whh, res_b, res_Wih, 644, WHH_RES, BIAS_RES, WT_RES);
  k_reorder_lstm<<<8, 256, 0, stream>>>(ts_Whh, ts_b, ts_Wih, 581, WHH_TS, BIAS_TS, WT_TS);

  auto mk = [&](const float* src, int sn, int sk, int k0, int kl,
                __bf16* dst, int NP, int KP, int ro) {
    int n = NP * KP;
    k_make_b<<<(n + 255) / 256, 256, 0, stream>>>(src, sn, sk, k0, kl, dst, NP, KP, ro);
  };
  mk(act_Wih, 2048, 580, 0, 516, B_ACTPRE, 2048, 544, 1);
  mk(res_Wih, 2048, 644, 0, 516, B_RESPRE, 2048, 544, 1);
  mk(ts_Wih,  2048, 581, 0, 516, B_TSPRE,  2048, 544, 1);
  mk(t2mean_W, 128, 516, 0, 516, B_T2MEAN, 128, 544, 0);
  mk(t2var_W,  128, 516, 0, 516, B_T2VAR,  128, 544, 0);
  mk(z2t_W,    516, 128, 0, 128, B_Z2T,    576, 128, 0);
  mk(attr_W1,  258, 516, 0, 516, B_ATTR1,  320, 544, 0);
  mk(attr_W2,    8, 258, 0, 258, B_ATTR2,   64, 288, 0);
  mk(e2act_W,   64, 512, 0, 512, B_E2ACT,   64, 512, 0);
  mk(e2res_W,   64, 512, 0, 512, B_E2RES,   64, 512, 0);
  mk(e2ts_W1,  256, 512, 0, 512, B_E2TS1,  256, 512, 0);

  k_pad_bias<<<1, 128, 0, stream>>>(t2mean_b, 128, PB_T2MEAN, 128);
  k_pad_bias<<<1, 128, 0, stream>>>(t2var_b, 128, PB_T2VAR, 128);
  k_pad_bias<<<3, 256, 0, stream>>>(z2t_b, 516, PB_Z2T, 576);
  k_pad_bias<<<2, 256, 0, stream>>>(attr_b1, 258, PB_ATTR1, 320);
  k_pad_bias<<<1, 64, 0, stream>>>(attr_b2, 8, PB_ATTR2, 64);
  k_pad_bias<<<1, 64, 0, stream>>>(e2act_b, 64, PB_E2ACT, 64);
  k_pad_bias<<<1, 64, 0, stream>>>(e2res_b, 64, PB_E2RES, 64);
  k_pad_bias<<<1, 256, 0, stream>>>(e2ts_b1, 256, PB_E2TS1, 256);

  k_zero<<<256, 256, 0, stream>>>((float*)HBUF, 8 * 512 * 512 / 2);
  k_zero<<<256, 256, 0, stream>>>(CBUF, 4 * 512 * 512);
  k_zero<<<256, 256, 0, stream>>>((float*)T_BF16, 512 * 544 / 2);
  k_zero<<<256, 256, 0, stream>>>((float*)TREC_BF16, 512 * 544 / 2);
  k_zero<<<256, 256, 0, stream>>>((float*)A1_BF16, 512 * 288 / 2);
  k_init_misc<<<2, 256, 0, stream>>>(attr0, acts, a2aW, a2ab, LENS, ACTIDX, RESIDX,
                                     TSPREV, T_BF16);

  const dim3 gL(8, 8), bL(128);
  // ---- encoder ----
  for (int t = 0; t < 64; ++t) {
    const __bf16* hin = (t & 1) ? H_ENC1 : H_ENC0;
    __bf16* hout = (t & 1) ? H_ENC0 : H_ENC1;
    k_lstm_step<<<gL, bL, 0, stream>>>(hin, WHH_ENC, BIAS_ENC, WT_ENC, nullptr,
                                       C_ENC, hout, acts, ress, tsin, t, 0,
                                       LENS, T_BF16);
  }

  // ---- latent + decoder precomputes ----
  k_gemm<<<dim3(8, 2), 128, 0, stream>>>(T_BF16, 544, B_T2MEAN, fout + MEAN_OFF, 128,
                                         nullptr, 0, PB_T2MEAN, 128, 544, 0);
  k_gemm<<<dim3(8, 2), 128, 0, stream>>>(T_BF16, 544, B_T2VAR, fout + LOGVAR_OFF, 128,
                                         nullptr, 0, PB_T2VAR, 128, 544, 0);
  k_z<<<256, 256, 0, stream>>>(fout + MEAN_OFF, fout + LOGVAR_OFF, eps, Z_BF16);
  k_gemm<<<dim3(8, 9), 128, 0, stream>>>(Z_BF16, 128, B_Z2T, nullptr, 0,
                                         TREC_BF16, 544, PB_Z2T, 516, 128, 1);
  k_gemm<<<dim3(8, 5), 128, 0, stream>>>(TREC_BF16, 544, B_ATTR1, nullptr, 0,
                                         A1_BF16, 288, PB_ATTR1, 258, 544, 1);
  k_gemm<<<dim3(8, 1), 128, 0, stream>>>(A1_BF16, 288, B_ATTR2, ATTR_TMP, 64,
                                         nullptr, 0, PB_ATTR2, 8, 288, 0);
  k_softmax8<<<512, 32, 0, stream>>>(ATTR_TMP, 64, fout + ATTR_OFF);
  k_gemm<<<dim3(8, 32), 128, 0, stream>>>(TREC_BF16, 544, B_ACTPRE, PRE_ACT, 2048,
                                          nullptr, 0, nullptr, 2048, 544, 0);
  k_gemm<<<dim3(8, 32), 128, 0, stream>>>(TREC_BF16, 544, B_RESPRE, PRE_RES, 2048,
                                          nullptr, 0, nullptr, 2048, 544, 0);
  k_gemm<<<dim3(8, 32), 128, 0, stream>>>(TREC_BF16, 544, B_TSPRE, PRE_TS, 2048,
                                          nullptr, 0, nullptr, 2048, 544, 0);

  // ---- decoder ----
  for (int t = 0; t < 64; ++t) {
    const __bf16* hina = (t & 1) ? H_ACT1 : H_ACT0;
    __bf16* houta = (t & 1) ? H_ACT0 : H_ACT1;
    k_lstm_step<<<gL, bL, 0, stream>>>(hina, WHH_ACT, BIAS_ACT, WT_ACT, PRE_ACT,
                                       C_ACT, houta, ACTIDX, nullptr, nullptr,
                                       t, 1, nullptr, nullptr);
    k_gemm<<<dim3(8, 1), 128, 0, stream>>>(houta, 512, B_E2ACT,
                                           fout + ACTS_OFF + (size_t)t * 64, 4096,
                                           nullptr, 0, PB_E2ACT, 64, 512, 0);
    k_argmax64<<<512, 64, 0, stream>>>(fout + ACTS_OFF + (size_t)t * 64, 4096, ACTIDX);

    const __bf16* hinr = (t & 1) ? H_RES1 : H_RES0;
    __bf16* houtr = (t & 1) ? H_RES0 : H_RES1;
    k_lstm_step<<<gL, bL, 0, stream>>>(hinr, WHH_RES, BIAS_RES, WT_RES, PRE_RES,
                                       C_RES, houtr, ACTIDX, RESIDX, nullptr,
                                       t, 2, nullptr, nullptr);
    k_gemm<<<dim3(8, 1), 128, 0, stream>>>(houtr, 512, B_E2RES,
                                           fout + RESS_OFF + (size_t)t * 64, 4096,
                                           nullptr, 0, PB_E2RES, 64, 512, 0);
    k_argmax64<<<512, 64, 0, stream>>>(fout + RESS_OFF + (size_t)t * 64, 4096, RESIDX);

    const __bf16* hint = (t & 1) ? H_TS1 : H_TS0;
    __bf16* houtt = (t & 1) ? H_TS0 : H_TS1;
    k_lstm_step<<<gL, bL, 0, stream>>>(hint, WHH_TS, BIAS_TS, WT_TS, PRE_TS,
                                       C_TS, houtt, ACTIDX, nullptr, TSPREV,
                                       t, 3, nullptr, nullptr);
    k_gemm<<<dim3(8, 4), 128, 0, stream>>>(houtt, 512, B_E2TS1, TSH, 256,
                                           nullptr, 0, PB_E2TS1, 256, 512, 1);
    k_ts_head2<<<512, 32, 0, stream>>>(TSH, e2ts_W2, e2ts_b2, fout + TS_OFF, TSPREV, t);
  }

  // ---- final softmaxes (in place on d_out) ----
  k_softmax64<<<32768, 64, 0, stream>>>(fout + ACTS_OFF);
  k_softmax64<<<32768, 64, 0, stream>>>(fout + RESS_OFF);
}
